// LSTM_Init_To_Many_43714177138896
// MI455X (gfx1250) — matrix operand order repaired and checked by reading, not compiled, not run on hardware
//
#include <hip/hip_runtime.h>
#include <hip/hip_bf16.h>

#define B_SZ   256
#define IN_SZ  256
#define H_SZ   1024
#define OUT_SZ 128
#define SEQ    150
#define KCH    (H_SZ / 32)   // 32 K-chunks of 32

typedef __bf16 bf16_t;
typedef bf16_t bf16x16 __attribute__((ext_vector_type(16)));
typedef bf16_t bf16x8  __attribute__((ext_vector_type(8)));
typedef float  f32x8   __attribute__((ext_vector_type(8)));

// ---------- helpers ----------

__device__ __forceinline__ bf16_t f2bf(float x) {
  unsigned u = __builtin_bit_cast(unsigned, x);
  unsigned r = u + 0x7FFFu + ((u >> 16) & 1u);   // round-to-nearest-even
  unsigned short s = (unsigned short)(r >> 16);
  return __builtin_bit_cast(bf16_t, s);
}

__device__ __forceinline__ float fast_sigmoid(float x) {
  return 1.0f / (1.0f + __expf(-x));
}
__device__ __forceinline__ float fast_tanh(float x) {
  return 2.0f / (1.0f + __expf(-2.0f * x)) - 1.0f;
}

// A fragment (16x32 bf16): ISA 7.12.2 16-bit A layout.
// lanes 0-15: M=lane, elems 0..7 -> K=k0..k0+7, elems 8..15 -> K=k0+16..+23
// lanes 16-31: M=lane-16, K ranges shifted by +8.
template <typename P>
__device__ __forceinline__ bf16x16 load_a_frag(P base, int m0, int k0, int lane, int ldk) {
  int row  = m0 + (lane & 15);
  int koff = (lane >> 4) * 8;
  auto p = base + row * ldk + k0 + koff;
  bf16x8 lo = *(const bf16x8*)p;
  bf16x8 hi = *(const bf16x8*)(p + 16);
  return __builtin_shufflevector(lo, hi, 0,1,2,3,4,5,6,7,8,9,10,11,12,13,14,15);
}

// B fragment (32x16 bf16): lane = column N (row-major weight row), lanes 0-15
// hold K=k0..k0+7, k0+16..k0+23, lanes 16-31 K shifted by +8. Two 16B loads.
__device__ __forceinline__ bf16x16 load_b_frag(const bf16_t* __restrict__ base,
                                               int n0, int k0, int lane, int ldk) {
  int col  = n0 + (lane & 15);
  int koff = (lane >> 4) * 8;
  const bf16_t* p = base + col * ldk + k0 + koff;
  bf16x8 lo = *(const bf16x8*)p;
  bf16x8 hi = *(const bf16x8*)(p + 16);
  return __builtin_shufflevector(lo, hi, 0,1,2,3,4,5,6,7,8,9,10,11,12,13,14,15);
}

#define WMMA_BF16(A, Bm, C) \
  __builtin_amdgcn_wmma_f32_16x16x32_bf16(false, (A), false, (Bm), (short)0, (C), false, false)

// ---------- prep kernels ----------

__global__ void cvt_f32_to_bf16(const float* __restrict__ src, bf16_t* __restrict__ dst, int n) {
  int i = blockIdx.x * blockDim.x + threadIdx.x;
  int stride = gridDim.x * blockDim.x;
  for (; i < n; i += stride) dst[i] = f2bf(src[i]);
}

__global__ void bias_sum(const float* __restrict__ a, const float* __restrict__ b,
                         float* __restrict__ o, int n) {
  int i = blockIdx.x * blockDim.x + threadIdx.x;
  if (i < n) o[i] = a[i] + b[i];
}

// h0 = x @ W_in_h^T + b_in_h (bf16), c0 = x @ W_in_c^T + b_in_c (f32). K=256: tiny.
__global__ void init_state(const float* __restrict__ x,
                           const float* __restrict__ Wh, const float* __restrict__ bh,
                           const float* __restrict__ Wc, const float* __restrict__ bc,
                           bf16_t* __restrict__ h0, float* __restrict__ c0) {
  int t = blockIdx.x * blockDim.x + threadIdx.x;   // [0, B*H)
  int b = t >> 10;
  int j = t & 1023;
  const float* xr = x  + b * IN_SZ;
  const float* wh = Wh + j * IN_SZ;
  const float* wc = Wc + j * IN_SZ;
  float sh = 0.f, sc = 0.f;
  #pragma unroll 4
  for (int k = 0; k < IN_SZ; k += 4) {
    float4 xv = *(const float4*)(xr + k);
    float4 hv = *(const float4*)(wh + k);
    float4 cv = *(const float4*)(wc + k);
    sh += xv.x * hv.x + xv.y * hv.y + xv.z * hv.z + xv.w * hv.w;
    sc += xv.x * cv.x + xv.y * cv.y + xv.z * cv.z + xv.w * cv.w;
  }
  h0[t] = f2bf(sh + bh[j]);
  c0[t] = sc + bc[j];
}

// ---------- fused per-step kernel ----------
// Blocks 0..63  : gates GEMM (step s) + LSTM cell update.
//   Each block owns 32 batch rows (async-staged to LDS once), each of its 8
//   waves owns 16 hidden columns x all 4 gates x 2 M-tiles (MT=2 halves the
//   L2 weight re-read vs one tile/wave). B fragments are double-buffered so
//   WMMAs overlap the next chunk's weight loads.
// Blocks 64..79 : output projection for step proj_s = s-1 (reads the same h).

__global__ __launch_bounds__(256) void lstm_fused(
    const bf16_t* __restrict__ h_in, bf16_t* __restrict__ h_out,
    float* __restrict__ c, const bf16_t* __restrict__ Whh,
    const float* __restrict__ bias4h,
    const bf16_t* __restrict__ Wout, const float* __restrict__ bout,
    float* __restrict__ out, int proj_s) {
  int lane = threadIdx.x & 31;
  int wid  = threadIdx.x >> 5;

  if (blockIdx.x < 64) {
    // ---------------- gates + cell update ----------------
    __shared__ __align__(64) bf16_t sA[32 * H_SZ];   // 64 KB: 32 rows of h
    int mg = blockIdx.x >> 3;                        // m-group: rows [mg*32, mg*32+32)

    // Async-stage the 32 shared batch rows into LDS (CDNA5 ASYNCcnt path).
    {
      const bf16_t* gsrc = h_in + mg * 32 * H_SZ;
      unsigned lbase = (unsigned)(size_t)(&sA[0]);
      int t = threadIdx.x;                           // 256 threads x 256 B each
      #pragma unroll
      for (int i = 0; i < 16; ++i) {
        unsigned loff = lbase + (unsigned)(t * 256 + i * 16);
        const bf16_t* g = gsrc + t * 128 + i * 8;
        asm volatile("global_load_async_to_lds_b128 %0, %1, off"
                     :: "v"(loff), "v"(g) : "memory");
      }
      asm volatile("s_wait_asynccnt 0x0" ::: "memory");
      __syncthreads();
    }

    int w  = (blockIdx.x << 3) + wid;   // global wave id, 0..511 (w>>6 == mg)
    int n0 = (w & 63) << 4;             // 64 hidden-column tiles

    f32x8 acc[2][4];
    #pragma unroll
    for (int t = 0; t < 2; ++t)
      #pragma unroll
      for (int g = 0; g < 4; ++g) acc[t][g] = f32x8{};

    bf16x16 bB[2][4];                   // double-buffered weight fragments
    #pragma unroll
    for (int g = 0; g < 4; ++g)
      bB[0][g] = load_b_frag(Whh, g * H_SZ + n0, 0, lane, H_SZ);

    #pragma unroll
    for (int kc = 0; kc < KCH; ++kc) {
      int cur = kc & 1, nxt = cur ^ 1;
      int k0 = kc * 32;
      if (kc + 1 < KCH) {
        #pragma unroll
        for (int g = 0; g < 4; ++g)
          bB[nxt][g] = load_b_frag(Whh, g * H_SZ + n0, k0 + 32, lane, H_SZ);
      }
      bf16x16 a0 = load_a_frag((const bf16_t*)sA, 0,  k0, lane, H_SZ);
      bf16x16 a1 = load_a_frag((const bf16_t*)sA, 16, k0, lane, H_SZ);
      #pragma unroll
      for (int g = 0; g < 4; ++g) {
        acc[0][g] = WMMA_BF16(a0, bB[cur][g], acc[0][g]);
        acc[1][g] = WMMA_BF16(a1, bB[cur][g], acc[1][g]);
      }
    }

    // Epilogue: C/D layout -> VGPR v, lanes 0-15 M=v, lanes 16-31 M=v+8; N=lane&15
    int nl = lane & 15, mh = lane >> 4;
    int n  = n0 + nl;
    float bi  = bias4h[n];
    float bf_ = bias4h[H_SZ + n];
    float bg  = bias4h[2 * H_SZ + n];
    float bo  = bias4h[3 * H_SZ + n];
    #pragma unroll
    for (int t = 0; t < 2; ++t) {
      int row_base = mg * 32 + t * 16 + mh * 8;
      #pragma unroll
      for (int v = 0; v < 8; ++v) {
        int idx = (row_base + v) * H_SZ + n;
        float ig = fast_sigmoid(acc[t][0][v] + bi);
        float fg = fast_sigmoid(acc[t][1][v] + bf_);
        float gg = fast_tanh(acc[t][2][v] + bg);
        float og = fast_sigmoid(acc[t][3][v] + bo);
        float cn = fg * c[idx] + ig * gg;
        c[idx] = cn;
        h_out[idx] = f2bf(og * fast_tanh(cn));
      }
    }
  } else {
    // ---------------- projection for previous step ----------------
    if (proj_s < 0) return;             // no h produced yet at s == 0
    int pb = blockIdx.x - 64;           // 0..15
    int w  = pb * 8 + wid;              // 0..127
    int m0 = (w >> 3) << 4;
    int n0 = (w & 7) << 4;

    f32x8 acc = {};
    bf16x16 bB[2];
    bB[0] = load_b_frag(Wout, n0, 0, lane, H_SZ);
    #pragma unroll
    for (int kc = 0; kc < KCH; ++kc) {
      int cur = kc & 1, nxt = cur ^ 1;
      int k0 = kc * 32;
      if (kc + 1 < KCH) bB[nxt] = load_b_frag(Wout, n0, k0 + 32, lane, H_SZ);
      bf16x16 a = load_a_frag(h_in, m0, k0, lane, H_SZ);
      acc = WMMA_BF16(a, bB[cur], acc);
    }

    int nl = lane & 15, mh = lane >> 4;
    int n  = n0 + nl;
    float bb = bout[n];
    #pragma unroll
    for (int v = 0; v < 8; ++v) {
      int row = m0 + mh * 8 + v;
      out[(row * SEQ + proj_s) * OUT_SZ + n] = acc[v] + bb;
    }
  }
}

// Standalone projection (used once, for the final step's h).
__global__ __launch_bounds__(256) void proj_step(
    const bf16_t* __restrict__ h_in, const bf16_t* __restrict__ Wout,
    const float* __restrict__ bout, float* __restrict__ out, int s) {
  int lane = threadIdx.x & 31;
  int w = blockIdx.x * (blockDim.x >> 5) + (threadIdx.x >> 5);
  int m0 = (w >> 3) << 4;
  int n0 = (w & 7) << 4;
  f32x8 acc = {};
  for (int k0 = 0; k0 < H_SZ; k0 += 32) {
    bf16x16 a = load_a_frag(h_in, m0, k0, lane, H_SZ);
    bf16x16 b = load_b_frag(Wout, n0, k0, lane, H_SZ);
    acc = WMMA_BF16(a, b, acc);
  }
  int nl = lane & 15, mh = lane >> 4;
  int n  = n0 + nl;
  float bb = bout[n];
  #pragma unroll
  for (int v = 0; v < 8; ++v) {
    int row = m0 + mh * 8 + v;
    out[(row * SEQ + s) * OUT_SZ + n] = acc[v] + bb;
  }
}

// ---------- launch ----------

extern "C" void kernel_launch(void* const* d_in, const int* in_sizes, int n_in,
                              void* d_out, int out_size, void* d_ws, size_t ws_size,
                              hipStream_t stream) {
  const float* init_input = (const float*)d_in[0];
  const float* W_in_h     = (const float*)d_in[1];
  const float* b_in_h     = (const float*)d_in[2];
  const float* W_in_c     = (const float*)d_in[3];
  const float* b_in_c     = (const float*)d_in[4];
  const float* W_hh       = (const float*)d_in[5];
  const float* b_ih       = (const float*)d_in[6];
  const float* b_hh       = (const float*)d_in[7];
  const float* W_out      = (const float*)d_in[8];
  const float* b_out      = (const float*)d_in[9];
  float* out = (float*)d_out;

  // Workspace layout (~10.8 MB, 256B-aligned offsets).
  char* ws = (char*)d_ws;
  bf16_t* Whh_bf  = (bf16_t*)(ws);                    // 4H*H bf16  = 8 MB (L2-resident)
  bf16_t* Wout_bf = (bf16_t*)(ws + 8388608);          // OUT*H bf16 = 256 KB
  float*  bias4h  = (float*)(ws + 8650752);           // 4H f32     = 16 KB
  float*  c_st    = (float*)(ws + 8667136);           // B*H f32    = 1 MB
  bf16_t* hbuf0   = (bf16_t*)(ws + 9715712);          // B*H bf16   = 512 KB
  bf16_t* hbuf1   = (bf16_t*)(ws + 10240000);         // B*H bf16   = 512 KB
  bf16_t* hb[2] = {hbuf0, hbuf1};

  // One-time prep.
  cvt_f32_to_bf16<<<512, 256, 0, stream>>>(W_hh,  Whh_bf,  4 * H_SZ * H_SZ);
  cvt_f32_to_bf16<<<64,  256, 0, stream>>>(W_out, Wout_bf, OUT_SZ * H_SZ);
  bias_sum<<<16, 256, 0, stream>>>(b_ih, b_hh, bias4h, 4 * H_SZ);
  init_state<<<(B_SZ * H_SZ) / 256, 256, 0, stream>>>(
      init_input, W_in_h, b_in_h, W_in_c, b_in_c, hb[0], c_st);

  // 150 fused launches: gates(s) [+ projection(s-1) in the same grid].
  for (int s = 0; s < SEQ; ++s) {
    lstm_fused<<<80, 256, 0, stream>>>(hb[s & 1], hb[(s + 1) & 1], c_st,
                                       Whh_bf, bias4h, Wout_bf, b_out, out, s - 1);
  }
  // Final projection for step 149 (its h lives in hb[0]).
  proj_step<<<16, 256, 0, stream>>>(hb[0], Wout_bf, b_out, out, SEQ - 1);
}
